// GRU_84980222919272
// MI455X (gfx1250) — compile-verified, matrix operand order checked by reading, not run on hardware
//
#include <hip/hip_runtime.h>

// ---------------------------------------------------------------------------
// 2-layer GRU (B=64, S=512, C=H=1024) for gfx1250 / MI455X, wave32 WMMA bf16.
//  - w_ih / w_hh converted once to bf16, pre-swizzled into WMMA B-fragment
//    layout (lane = N column, 16 contiguous K per lane).
//  - gi = x @ w_ih^T + b_ih hoisted into one big batched WMMA GEMM per layer.
//  - per-step kernel: gh = h @ w_hh^T via 3 fused gate WMMAs per K-step,
//    A fragment staged in LDS (f32 state -> bf16 fragments), gate math fused.
// ---------------------------------------------------------------------------

typedef __attribute__((ext_vector_type(16))) __bf16 v16bf;
typedef __attribute__((ext_vector_type(8)))  __bf16 v8bf;
typedef __attribute__((ext_vector_type(8)))  float  v8f;

#define B_   64
#define S_   512
#define C_   1024
#define H_   1024
#define G3H  3072
#define KT   32                 // K tiles: 1024 / 32
#define NT   192                // N tiles: 3072 / 16
#define WMAT ((size_t)G3H * C_) // elements per weight matrix

// ---------------------------------------------------------------------------
// One-time weight swizzle: f32 [3072][1024] row-major -> bf16 B-fragments.
// Fragment tile (nt, kt): 32 lanes x 16 elems; lane L holds column
// n = nt*16 + (L&15), elems e cover k = kt*32 + (L>>4)*16 + e.
// Linear element index inside a matrix: ((nt*KT + kt)*32 + lane)*16 + e.
// ---------------------------------------------------------------------------
__global__ void __launch_bounds__(256)
swizzle_weights(const float* __restrict__ w_ih,
                const float* __restrict__ w_hh,
                __bf16* __restrict__ out)
{
  int gid = blockIdx.x * 256 + threadIdx.x;     // one 8-elem (16B) chunk each
  const int cpm = NT * KT * 64;                 // chunks per matrix
  int mat = gid / cpm;
  int c   = gid % cpm;                          // c*8 == linear element index
  int nt  = c / (KT * 64);
  int rem = c % (KT * 64);
  int kt  = rem >> 6;
  int lane = (rem >> 1) & 31;
  int h2   = rem & 1;
  int n  = nt * 16 + (lane & 15);
  int k0 = kt * 32 + (lane >> 4) * 16 + h2 * 8;
  const float* src = (mat == 0) ? w_ih
                   : (mat == 1) ? w_hh
                   : (mat == 2) ? (w_ih + WMAT)
                                : (w_hh + WMAT);
  const float4* p = (const float4*)(src + (size_t)n * C_ + k0);
  float4 f0 = p[0], f1 = p[1];
  v8bf v;
  v[0]=(__bf16)f0.x; v[1]=(__bf16)f0.y; v[2]=(__bf16)f0.z; v[3]=(__bf16)f0.w;
  v[4]=(__bf16)f1.x; v[5]=(__bf16)f1.y; v[6]=(__bf16)f1.z; v[7]=(__bf16)f1.w;
  *(v8bf*)(out + (size_t)mat * WMAT + (size_t)c * 8) = v;
}

// ---------------------------------------------------------------------------
// Stage one A m-tile (16 rows x K=1024) from f32 source into LDS bf16
// fragments. A fragment layout (16-bit A 16x32): lane L holds row m = L&15,
// elems e: k = (L>>4)*16 + e  -> 16 contiguous k (same order as B).
// LDS linear layout: (kt*32 + lane)*16 + e, total 32 KB.
// Row addressing: row m -> Abase + tloc*strideT + (b0+m)*strideB.
// ---------------------------------------------------------------------------
__device__ __forceinline__ void stage_A(const float* __restrict__ Abase,
                                        size_t strideT, size_t strideB,
                                        int mt, __bf16* lds)
{
  const int tloc = mt >> 2;
  const int b0   = (mt & 3) * 16;
  const float* base = Abase + (size_t)tloc * strideT;
  #pragma unroll
  for (int i = 0; i < 8; ++i) {
    int c    = threadIdx.x + i * 256;   // 0..2047, c*8 = lds element offset
    int lane = (c >> 1) & 31;
    int h2   = c & 1;
    int kt   = c >> 6;
    int m    = lane & 15;
    int k0   = kt * 32 + (lane >> 4) * 16 + h2 * 8;
    const float4* p = (const float4*)(base + (size_t)(b0 + m) * strideB + k0);
    float4 f0 = p[0], f1 = p[1];
    v8bf v;
    v[0]=(__bf16)f0.x; v[1]=(__bf16)f0.y; v[2]=(__bf16)f0.z; v[3]=(__bf16)f0.w;
    v[4]=(__bf16)f1.x; v[5]=(__bf16)f1.y; v[6]=(__bf16)f1.z; v[7]=(__bf16)f1.w;
    *(v8bf*)(lds + (size_t)c * 8) = v;
  }
}

// ---------------------------------------------------------------------------
// Big batched GEMM: out[rows][3072] = A[rows][1024] @ W^T + bias.
// rows = S*B grouped as (t, b); row tile mt = t*4 + b/16.
// One wave -> one 16x16 output tile, kt-loop of 32 WMMAs.
// Block = 8 waves sharing one A m-tile via LDS.
// ---------------------------------------------------------------------------
__global__ void __launch_bounds__(256)
gemm_gi(const float* __restrict__ A, size_t strideT, size_t strideB,
        const __bf16* __restrict__ W,      // swizzled fragments
        const float* __restrict__ bias,    // [3072]
        float* __restrict__ out)           // [rows][3072]
{
  __shared__ __bf16 ldsA[KT * 512];
  const int w    = threadIdx.x >> 5;
  const int lane = threadIdx.x & 31;
  const int mt   = blockIdx.x / 24;          // 24 blocks per m-tile (192/8)
  const int nt   = (blockIdx.x % 24) * 8 + w;

  stage_A(A, strideT, strideB, mt, ldsA);
  __syncthreads();

  v8f acc = {};
  const v16bf* Bv = (const v16bf*)W;
  #pragma unroll 4
  for (int kt = 0; kt < KT; ++kt) {
    v16bf a = *(const v16bf*)(ldsA + (size_t)(kt * 32 + lane) * 16);
    v16bf b = Bv[(size_t)(nt * KT + kt) * 32 + lane];
    acc = __builtin_amdgcn_wmma_f32_16x16x32_bf16(false, a, false, b,
                                                  (short)0, acc, false, false);
  }

  const int nl = lane & 15, hi = lane >> 4;
  const int ng = nt * 16 + nl;
  const float bv = bias[ng];
  #pragma unroll
  for (int r = 0; r < 8; ++r) {
    int m = r + hi * 8;  // C/D layout: VGPR r holds M=r (lo lanes) / r+8 (hi)
    out[(size_t)(mt * 16 + m) * G3H + ng] = acc[r] + bv;
  }
}

__device__ __forceinline__ float sigmoidf_(float x)
{
  return 1.0f / (1.0f + __expf(-x));
}

// ---------------------------------------------------------------------------
// One GRU time step (one layer):
//   gh = h_prev @ w_hh^T  (3 gate WMMAs per K-step, A shared in LDS)
//   r = sig(gi_r + gh_r + bhh_r), z = sig(gi_z + gh_z + bhh_z)
//   n = tanh(gi_n + r*(gh_n + bhh_n));  h = (1-z)*n + z*h_prev
// Grid: 32 blocks x 256 thr; block = (mt = bid>>3, 8 n-tiles).
// ---------------------------------------------------------------------------
__global__ void __launch_bounds__(256)
gru_step(const float* __restrict__ gi_t,    // [64][3072] (includes b_ih)
         const float* __restrict__ h_prev,  // [64][1024] f32 (unused if first)
         const __bf16* __restrict__ Whh,    // swizzled fragments
         const float* __restrict__ bhh,     // [3072]
         float* __restrict__ h_out,         // [64][1024] f32
         int first)
{
  __shared__ __bf16 ldsA[KT * 512];
  const int w    = threadIdx.x >> 5;
  const int lane = threadIdx.x & 31;
  const int mt   = blockIdx.x >> 3;            // 0..3
  const int nt   = (blockIdx.x & 7) * 8 + w;   // 0..63 (H tile)

  v8f accR = {}, accZ = {}, accN = {};
  if (!first) {
    stage_A(h_prev, 0, (size_t)H_, mt, ldsA);
    __syncthreads();
    const v16bf* Bv = (const v16bf*)Whh;
    const int ntR = nt, ntZ = nt + 64, ntN = nt + 128;
    #pragma unroll 2
    for (int kt = 0; kt < KT; ++kt) {
      v16bf a  = *(const v16bf*)(ldsA + (size_t)(kt * 32 + lane) * 16);
      v16bf br = Bv[(size_t)(ntR * KT + kt) * 32 + lane];
      v16bf bz = Bv[(size_t)(ntZ * KT + kt) * 32 + lane];
      v16bf bn = Bv[(size_t)(ntN * KT + kt) * 32 + lane];
      __builtin_prefetch(&Bv[(size_t)(ntR * KT + kt + 4) * 32 + lane], 0, 1);
      accR = __builtin_amdgcn_wmma_f32_16x16x32_bf16(false, a, false, br,
                                                     (short)0, accR, false, false);
      accZ = __builtin_amdgcn_wmma_f32_16x16x32_bf16(false, a, false, bz,
                                                     (short)0, accZ, false, false);
      accN = __builtin_amdgcn_wmma_f32_16x16x32_bf16(false, a, false, bn,
                                                     (short)0, accN, false, false);
    }
  }

  const int nl = lane & 15, hi = lane >> 4;
  const int ng = nt * 16 + nl;                 // column in H
  const float bR = bhh[ng], bZ = bhh[H_ + ng], bN = bhh[2 * H_ + ng];
  #pragma unroll
  for (int r = 0; r < 8; ++r) {
    const int mg = mt * 16 + r + hi * 8;       // batch row
    const float* gir = gi_t + (size_t)mg * G3H;
    float iR = gir[ng], iZ = gir[H_ + ng], iN = gir[2 * H_ + ng];
    float hR = accR[r] + bR;
    float hZ = accZ[r] + bZ;
    float hN = accN[r] + bN;
    float rg = sigmoidf_(iR + hR);
    float zg = sigmoidf_(iZ + hZ);
    float ngate = tanhf(iN + rg * hN);
    float hp = first ? 0.0f : h_prev[(size_t)mg * H_ + ng];
    h_out[(size_t)mg * H_ + ng] = (1.0f - zg) * ngate + zg * hp;
  }
}

// ---------------------------------------------------------------------------
// Host driver. Workspace layout (bytes):
//   [0)                 bf16 weights, 4 matrices       25,165,824
//   [+25,165,824)       gi   f32 [32768][3072]        402,653,184 (reused L0/L1)
//   [+427,819,008)      h0   f32 [512][64][1024]      134,217,728
//   [+562,036,736)      h1 ping/pong f32 2x[64][1024]     524,288
// ---------------------------------------------------------------------------
extern "C" void kernel_launch(void* const* d_in, const int* in_sizes, int n_in,
                              void* d_out, int out_size, void* d_ws, size_t ws_size,
                              hipStream_t stream)
{
  const float* x   = (const float*)d_in[0];   // [64][512][1024]
  const float* wih = (const float*)d_in[1];   // [2][3072][1024]
  const float* whh = (const float*)d_in[2];   // [2][3072][1024]
  const float* bih = (const float*)d_in[3];   // [2][3072]
  const float* bhh = (const float*)d_in[4];   // [2][3072]

  char*   ws  = (char*)d_ws;
  __bf16* wbf = (__bf16*)ws;
  size_t off = (size_t)4 * WMAT * sizeof(__bf16);
  float* gi  = (float*)(ws + off); off += (size_t)S_ * B_ * G3H * sizeof(float);
  float* h0  = (float*)(ws + off); off += (size_t)S_ * B_ * H_ * sizeof(float);
  float* h1a = (float*)(ws + off); off += (size_t)B_ * H_ * sizeof(float);
  float* h1b = (float*)(ws + off);

  // 1) one-time weight swizzle to bf16 fragments (resident in L2 afterwards)
  {
    const int chunks = 4 * NT * KT * 64;      // 1,572,864
    swizzle_weights<<<chunks / 256, 256, 0, stream>>>(wih, whh, wbf);
  }

  const int gemm_blocks = (S_ * B_ / 16) * (NT / 8);   // 2048 * 24 = 49152

  // 2) layer-0 input GEMM: gi = x @ w_ih0^T + b_ih0   (x is [b][t][k])
  gemm_gi<<<gemm_blocks, 256, 0, stream>>>(
      x, (size_t)C_, (size_t)S_ * C_, wbf, bih, gi);

  // 3) layer-0 recurrence; h0[t] doubles as layer-1 input history
  for (int t = 0; t < S_; ++t) {
    gru_step<<<32, 256, 0, stream>>>(
        gi + (size_t)t * B_ * G3H,
        (t == 0) ? h0 : h0 + (size_t)(t - 1) * B_ * H_,
        wbf + WMAT,              // w_hh layer 0
        bhh,
        h0 + (size_t)t * B_ * H_,
        t == 0 ? 1 : 0);
  }

  // 4) layer-1 input GEMM over full h0 history (h0 is [t][b][k])
  gemm_gi<<<gemm_blocks, 256, 0, stream>>>(
      h0, (size_t)B_ * H_, (size_t)H_, wbf + 2 * WMAT, bih + G3H, gi);

  // 5) layer-1 recurrence with ping-pong state; final step writes d_out
  for (int t = 0; t < S_; ++t) {
    float* cur  = (t & 1) ? h1b : h1a;
    float* prev = (t & 1) ? h1a : h1b;
    if (t == S_ - 1) cur = (float*)d_out;
    gru_step<<<32, 256, 0, stream>>>(
        gi + (size_t)t * B_ * G3H,
        (t == 0) ? h1a : prev,
        wbf + 3 * WMAT,          // w_hh layer 1
        bhh + G3H,
        cur,
        t == 0 ? 1 : 0);
  }
}
